// SelfAttention_60129542948
// MI455X (gfx1250) — matrix operand order repaired and checked by reading, not compiled, not run on hardware
//
#include <hip/hip_runtime.h>
#include <hip/hip_bf16.h>

#define B_  16
#define C_  512
#define CK_ 64
#define L_  2048

typedef __attribute__((ext_vector_type(16))) __bf16 bf16x16;
typedef __attribute__((ext_vector_type(8)))  float  floatx8;
typedef unsigned int uint32x4 __attribute__((ext_vector_type(4)));
typedef int          int32x8  __attribute__((ext_vector_type(8)));
typedef int          int32x4  __attribute__((ext_vector_type(4)));

union Frag { bf16x16 v; uint4 u[2]; };

__device__ inline floatx8 wmma_bf16(bf16x16 a, bf16x16 b, floatx8 c) {
  return __builtin_amdgcn_wmma_f32_16x16x32_bf16(false, a, false, b, (short)0, c,
                                                 false, false);
}

// f32 -> bf16 round-to-nearest-even
__device__ inline ushort f2bf(float x) {
  union { float f; unsigned u; } v; v.f = x;
  unsigned r = v.u + 0x7fffu + ((v.u >> 16) & 1u);
  return (ushort)(r >> 16);
}

// A fragment (16x32, row-major MxK, ld in elements).
// Lane layout: m = lane%16, chunks K=[h*8, h*8+8) and K=[h*8+16, h*8+24), h=lane/16.
__device__ inline bf16x16 load_a_g(const ushort* p, int ld) {
  int lane = threadIdx.x & 31;
  const ushort* row = p + (size_t)(lane & 15) * ld + ((lane >> 4) << 3);
  Frag f;
  f.u[0] = *(const uint4*)(row);
  f.u[1] = *(const uint4*)(row + 16);
  return f.v;
}

// ---------------------------------------------------------------------------
// Tensor Data Mover: async-load a 2D tile (rows x 64 bf16) into LDS with
// row padding 64 -> 72 elements (32 DWORDs data + 4 DWORDs pad).
// Descriptor per CDNA5 ISA.
// This toolchain exposes the 6-arg builtin form:
//   (v4u group0, v8i group1, v4i group2, v4i group3, v8i extra, i32 cpol)
// ---------------------------------------------------------------------------
__device__ inline void tdm_load_tile64(const ushort* gsrc, unsigned lds_off,
                                       unsigned rows) {
  unsigned long long ga = (unsigned long long)(uintptr_t)gsrc;
  uint32x4 g0;
  g0[0] = 1u;                                   // count=1, user mode
  g0[1] = lds_off;                              // lds_addr (bytes)
  g0[2] = (unsigned)(ga & 0xffffffffu);         // global_addr[31:0]
  g0[3] = (unsigned)((ga >> 32) & 0x01ffffffu)  // global_addr[56:32]
          | (2u << 30);                         // type = 2 ("image")
  int32x8 g1;
  g1[0] = (int)((1u << 16)     // data_size = 1 -> 2 bytes
                | (1u << 20)   // pad_enable
                | (4u << 22)   // pad_interval: code 4 -> every 32 DWORDs (128B)
                | (3u << 25)); // pad_amount:   code 3 -> 4 DWORDs (16B)
  g1[1] = (int)(64u << 16);    // tensor_dim0 = 64 (bits 79:48, low half)
  g1[2] = (int)(rows << 16);   // tensor_dim0 hi=0 | tensor_dim1[15:0] = rows
  g1[3] = (int)(64u << 16);    // tensor_dim1 hi=0 | tile_dim0 = 64
  g1[4] = (int)rows;           // tile_dim1 = rows, tile_dim2 = 0
  g1[5] = 64;                  // tensor_dim0_stride[31:0] = 64 elements
  g1[6] = 0;                   // stride0 hi | tensor_dim1_stride lo
  g1[7] = 0;
  int32x4 z4 = {0, 0, 0, 0};               // groups 2/3: zero (2D tensor)
  int32x8 z8i = {0, 0, 0, 0, 0, 0, 0, 0};  // extra group: zero
  __builtin_amdgcn_tensor_load_to_lds(g0, g1, z4, z4, z8i, 0);
}

// ---------------------------------------------------------------------------
// f32 -> bf16 bulk conversion
// ---------------------------------------------------------------------------
__global__ void cvt_kernel(const float* __restrict__ in, ushort* __restrict__ out,
                           int n4) {
  int i = blockIdx.x * blockDim.x + threadIdx.x;
  if (i < n4) {
    float4 f = ((const float4*)in)[i];
    ushort4 o;
    o.x = f2bf(f.x); o.y = f2bf(f.y); o.z = f2bf(f.z); o.w = f2bf(f.w);
    ((ushort4*)out)[i] = o;
  }
}

// ---------------------------------------------------------------------------
// Projection GEMM: Out(M x L) = W(M x 512) @ x_b(512 x L) + bias
// transposed==1: store as (L, M) per batch (for q/k, M=64)
// transposed==0: store as (M, L) per batch (for v, M=512)
// Block tile 64(M) x 128(N), 8 waves in 2x4, K stepped by 32 through LDS x^T.
// ---------------------------------------------------------------------------
__global__ __launch_bounds__(256) void proj_kernel(
    const ushort* __restrict__ W, const float* __restrict__ bias,
    const ushort* __restrict__ xbf, ushort* __restrict__ out,
    int M, int transposed) {
  __shared__ __align__(16) ushort xT[128][40];  // (i, c) transposed tile, padded

  const int b     = blockIdx.z;
  const int m_blk = blockIdx.y * 64;
  const int n_blk = blockIdx.x * 128;
  const ushort* xB = xbf + (size_t)b * C_ * L_;

  const int t = threadIdx.x;
  const int wave = t >> 5, lane = t & 31;
  const int wm = wave >> 2, wn = wave & 3;
  const int m0 = m_blk + wm * 32;
  const int n0 = wn * 32;  // LDS-local

  floatx8 acc[2][2];
  const floatx8 z8 = {0.f, 0.f, 0.f, 0.f, 0.f, 0.f, 0.f, 0.f};
  for (int i = 0; i < 2; ++i)
    for (int j = 0; j < 2; ++j) acc[i][j] = z8;

  const int c_local = t >> 3;        // 0..31
  const int i_local = (t & 7) * 16;  // 0..112

  for (int ks = 0; ks < C_ / 32; ++ks) {
    __syncthreads();
    {  // stage x^T tile: global rows (c, i-contig) -> LDS (i, c)
      const ushort* src = xB + (size_t)(ks * 32 + c_local) * L_ + n_blk + i_local;
      union { uint4 u; ushort s[8]; } d0, d1;
      d0.u = *(const uint4*)src;
      d1.u = *(const uint4*)(src + 8);
#pragma unroll
      for (int e = 0; e < 8; ++e) {
        xT[i_local + e][c_local]     = d0.s[e];
        xT[i_local + 8 + e][c_local] = d1.s[e];
      }
    }
    __syncthreads();

    bf16x16 a0 = load_a_g(W + (size_t)m0 * C_ + ks * 32, C_);
    bf16x16 a1 = load_a_g(W + (size_t)(m0 + 16) * C_ + ks * 32, C_);

    Frag b0, b1;
    {
      int n = lane & 15, k0 = (lane >> 4) << 3;
      b0.u[0] = *(const uint4*)&xT[n0 + n][k0];
      b0.u[1] = *(const uint4*)&xT[n0 + n][k0 + 16];
      b1.u[0] = *(const uint4*)&xT[n0 + 16 + n][k0];
      b1.u[1] = *(const uint4*)&xT[n0 + 16 + n][k0 + 16];
    }
    acc[0][0] = wmma_bf16(a0, b0.v, acc[0][0]);
    acc[0][1] = wmma_bf16(a0, b1.v, acc[0][1]);
    acc[1][0] = wmma_bf16(a1, b0.v, acc[1][0]);
    acc[1][1] = wmma_bf16(a1, b1.v, acc[1][1]);
  }

  const int n = lane & 15, h = lane >> 4;
#pragma unroll
  for (int mi = 0; mi < 2; ++mi)
#pragma unroll
    for (int nj = 0; nj < 2; ++nj) {
      int ckb = m0 + mi * 16 + h * 8;       // base output row (8 contiguous via r)
      int ig  = n_blk + n0 + nj * 16 + n;   // output column (position)
      if (transposed) {
        union { uint4 u; ushort s[8]; } pk;
#pragma unroll
        for (int r = 0; r < 8; ++r)
          pk.s[r] = f2bf(acc[mi][nj][r] + bias[ckb + r]);
        *(uint4*)(out + ((size_t)b * L_ + ig) * M + ckb) = pk.u;
      } else {
#pragma unroll
        for (int r = 0; r < 8; ++r)
          out[((size_t)b * M + ckb + r) * L_ + ig] =
              f2bf(acc[mi][nj][r] + bias[ckb + r]);
      }
    }
}

// ---------------------------------------------------------------------------
// Fused flash attention: per block = 32 query rows x all 512 channels.
// 8 waves; wave w owns channels [w*64, w*64+64). j streamed in tiles of 32.
// K tiles delivered by the Tensor Data Mover (wave 0 issues, TENSORcnt wait).
// ---------------------------------------------------------------------------
__global__ __launch_bounds__(256) void attn_kernel(
    const ushort* __restrict__ qb,   // (B, L, 64)
    const ushort* __restrict__ kb,   // (B, L, 64)
    const ushort* __restrict__ vb,   // (B, 512, L)
    const float*  __restrict__ xres, // (B, 512, L) f32 residual
    const float*  __restrict__ gamma,
    float* __restrict__ out) {
  __shared__ __align__(16) ushort kT[32][72];     // K tile (j, ck), padded rows
  __shared__ __align__(16) ushort P[8][32][40];   // per-wave P tile (i, j)

  const int b  = blockIdx.y;
  const int i0 = blockIdx.x * 32;
  const int t  = threadIdx.x;
  const int wave = t >> 5, lane = t & 31;
  const int c0 = wave * 64;
  const int n = lane & 15, h = lane >> 4;
  const unsigned kT_lds = (unsigned)(uintptr_t)(&kT[0][0]);

  // Q fragments live in registers for the whole j loop.
  bf16x16 qf[2][2];
#pragma unroll
  for (int mi = 0; mi < 2; ++mi)
#pragma unroll
    for (int kk = 0; kk < 2; ++kk)
      qf[mi][kk] =
          load_a_g(qb + ((size_t)b * L_ + i0 + mi * 16) * CK_ + kk * 32, CK_);

  const floatx8 z8 = {0.f, 0.f, 0.f, 0.f, 0.f, 0.f, 0.f, 0.f};
  floatx8 o[2][4];
#pragma unroll
  for (int mi = 0; mi < 2; ++mi)
#pragma unroll
    for (int nc = 0; nc < 4; ++nc) o[mi][nc] = z8;

  float mst[2][8], lst[2][8];
#pragma unroll
  for (int mi = 0; mi < 2; ++mi)
#pragma unroll
    for (int r = 0; r < 8; ++r) { mst[mi][r] = -1e30f; lst[mi][r] = 0.f; }

  const int srow  = t >> 3;        // 0..31 (prefetch address pattern)
  const int skoff = (t & 7) * 8;

  for (int j0 = 0; j0 < L_; j0 += 32) {
    __syncthreads();   // everyone done reading previous kT
    if (wave == 0) {   // TDM: DMA the 32x64 K tile into padded LDS rows
      tdm_load_tile64(kb + ((size_t)b * L_ + j0) * CK_, kT_lds, 32u);
      __builtin_amdgcn_s_wait_tensorcnt(0);
    }
    __syncthreads();   // tile visible to all waves
    if (j0 + 32 < L_) {  // prefetch next K tile into the cache hierarchy
      __builtin_prefetch(kb + ((size_t)b * L_ + j0 + 32 + srow) * CK_ + skoff, 0, 1);
      __builtin_prefetch(vb + ((size_t)b * C_ + c0 + (t & 63)) * L_ + j0 + 32, 0, 1);
    }

    // ---- S = Q K^T (32x32, K=64) ----
    floatx8 s[2][2];
#pragma unroll
    for (int mi = 0; mi < 2; ++mi)
#pragma unroll
      for (int nj = 0; nj < 2; ++nj) s[mi][nj] = z8;

    Frag kf[2][2];
    {
      int k0 = h << 3;
#pragma unroll
      for (int nj = 0; nj < 2; ++nj) {
        kf[nj][0].u[0] = *(const uint4*)&kT[nj * 16 + n][k0];
        kf[nj][0].u[1] = *(const uint4*)&kT[nj * 16 + n][k0 + 16];
        kf[nj][1].u[0] = *(const uint4*)&kT[nj * 16 + n][32 + k0];
        kf[nj][1].u[1] = *(const uint4*)&kT[nj * 16 + n][32 + k0 + 16];
      }
    }
#pragma unroll
    for (int mi = 0; mi < 2; ++mi)
#pragma unroll
      for (int nj = 0; nj < 2; ++nj) {
        s[mi][nj] = wmma_bf16(qf[mi][0], kf[nj][0].v, s[mi][nj]);
        s[mi][nj] = wmma_bf16(qf[mi][1], kf[nj][1].v, s[mi][nj]);
      }

    // ---- online softmax (row = r + 8h per mi-tile, 16 lanes per row) ----
#pragma unroll
    for (int mi = 0; mi < 2; ++mi)
#pragma unroll
      for (int r = 0; r < 8; ++r) {
        float mx = fmaxf(s[mi][0][r], s[mi][1][r]);
#pragma unroll
        for (int off = 8; off >= 1; off >>= 1)
          mx = fmaxf(mx, __shfl_xor(mx, off, 32));
        float mnew = fmaxf(mst[mi][r], mx);
        float corr = __expf(mst[mi][r] - mnew);
        mst[mi][r] = mnew;
        float p0 = __expf(s[mi][0][r] - mnew);
        float p1 = __expf(s[mi][1][r] - mnew);
        s[mi][0][r] = p0; s[mi][1][r] = p1;
        float ps = p0 + p1;
#pragma unroll
        for (int off = 8; off >= 1; off >>= 1) ps += __shfl_xor(ps, off, 32);
        lst[mi][r] = lst[mi][r] * corr + ps;
#pragma unroll
        for (int nc = 0; nc < 4; ++nc) o[mi][nc][r] *= corr;
      }

    // ---- P: D-layout -> per-wave LDS (row-major i x j) ----
#pragma unroll
    for (int mi = 0; mi < 2; ++mi)
#pragma unroll
      for (int nj = 0; nj < 2; ++nj)
#pragma unroll
        for (int r = 0; r < 8; ++r)
          P[wave][mi * 16 + r + 8 * h][nj * 16 + n] = f2bf(s[mi][nj][r]);

    // ---- O += P @ V^T  (A from LDS, B streamed straight from global v) ----
    bf16x16 pa[2];
    {
      int k0 = h << 3;
#pragma unroll
      for (int mi = 0; mi < 2; ++mi) {
        Frag f;
        f.u[0] = *(const uint4*)&P[wave][mi * 16 + n][k0];
        f.u[1] = *(const uint4*)&P[wave][mi * 16 + n][k0 + 16];
        pa[mi] = f.v;
      }
    }
#pragma unroll
    for (int nc = 0; nc < 4; ++nc) {
      Frag vf;
      const ushort* vrow =
          vb + ((size_t)b * C_ + c0 + nc * 16 + n) * L_ + j0 + (h << 3);
      vf.u[0] = *(const uint4*)vrow;
      vf.u[1] = *(const uint4*)(vrow + 16);
#pragma unroll
      for (int mi = 0; mi < 2; ++mi)
        o[mi][nc] = wmma_bf16(pa[mi], vf.v, o[mi][nc]);
    }
  }

  // ---- epilogue: normalize, gamma * O + x ----
  const float g = gamma[0];
#pragma unroll
  for (int mi = 0; mi < 2; ++mi)
#pragma unroll
    for (int nc = 0; nc < 4; ++nc)
#pragma unroll
      for (int r = 0; r < 8; ++r) {
        int ig = i0 + mi * 16 + r + 8 * h;
        int cg = c0 + nc * 16 + n;
        size_t idx = ((size_t)b * C_ + cg) * L_ + ig;
        out[idx] = g * (o[mi][nc][r] / lst[mi][r]) + xres[idx];
      }
}

// ---------------------------------------------------------------------------
extern "C" void kernel_launch(void* const* d_in, const int* in_sizes, int n_in,
                              void* d_out, int out_size, void* d_ws,
                              size_t ws_size, hipStream_t stream) {
  const float* x     = (const float*)d_in[0];
  const float* Wq    = (const float*)d_in[1];
  const float* bq    = (const float*)d_in[2];
  const float* Wk    = (const float*)d_in[3];
  const float* bk    = (const float*)d_in[4];
  const float* Wv    = (const float*)d_in[5];
  const float* bv    = (const float*)d_in[6];
  const float* gamma = (const float*)d_in[7];
  float* out = (float*)d_out;

  char* ws = (char*)d_ws;
  size_t off = 0;
  auto alloc = [&](size_t bytes) {
    char* p = ws + off;
    off = (off + bytes + 255) & ~(size_t)255;
    return p;
  };
  ushort* xbf  = (ushort*)alloc((size_t)B_ * C_ * L_ * 2);
  ushort* wqb  = (ushort*)alloc((size_t)CK_ * C_ * 2);
  ushort* wkb  = (ushort*)alloc((size_t)CK_ * C_ * 2);
  ushort* wvb  = (ushort*)alloc((size_t)C_ * C_ * 2);
  ushort* qbuf = (ushort*)alloc((size_t)B_ * L_ * CK_ * 2);
  ushort* kbuf = (ushort*)alloc((size_t)B_ * L_ * CK_ * 2);
  ushort* vbuf = (ushort*)alloc((size_t)B_ * C_ * L_ * 2);

  auto cvt = [&](const float* in, ushort* o, int nelem) {
    int n4 = nelem / 4;
    cvt_kernel<<<(n4 + 255) / 256, 256, 0, stream>>>(in, o, n4);
  };
  cvt(x, xbf, B_ * C_ * L_);
  cvt(Wq, wqb, CK_ * C_);
  cvt(Wk, wkb, CK_ * C_);
  cvt(Wv, wvb, C_ * C_);

  dim3 gqk(L_ / 128, 1, B_);
  proj_kernel<<<gqk, 256, 0, stream>>>(wqb, bq, xbf, qbuf, CK_, 1);
  proj_kernel<<<gqk, 256, 0, stream>>>(wkb, bk, xbf, kbuf, CK_, 1);
  dim3 gv(L_ / 128, C_ / 64, B_);
  proj_kernel<<<gv, 256, 0, stream>>>(wvb, bv, xbf, vbuf, C_, 0);

  dim3 ga(L_ / 32, B_);
  attn_kernel<<<ga, 256, 0, stream>>>(qbuf, kbuf, vbuf, x, gamma, out);
}
